// BlockCirculantLinear_82995948028093
// MI455X (gfx1250) — matrix operand order repaired and checked by reading, not compiled, not run on hardware
//
#include <hip/hip_runtime.h>
#include <hip/hip_bf16.h>
#include <stdint.h>

typedef __attribute__((ext_vector_type(16))) __bf16 v16bf;
typedef __attribute__((ext_vector_type(8)))  float  v8f;
typedef unsigned int u32x4 __attribute__((ext_vector_type(4)));
typedef int          i32x8 __attribute__((ext_vector_type(8)));
typedef int          i32x4 __attribute__((ext_vector_type(4)));

union Frag16 { v16bf v; uint4 q[2]; };

#ifndef USE_TDM
#define USE_TDM 1
#endif
#if USE_TDM && __has_builtin(__builtin_amdgcn_tensor_load_to_lds) && \
    __has_builtin(__builtin_amdgcn_s_wait_tensorcnt)
#define TDM_PATH 1
#else
#define TDM_PATH 0
#endif

__device__ __forceinline__ unsigned int f2bf(float f) {
    unsigned int u = __float_as_uint(f);
    return (u + 0x7fffu + ((u >> 16) & 1u)) >> 16;   // RNE
}
__device__ __forceinline__ unsigned int packbf(float lo, float hi) {
    return f2bf(lo) | (f2bf(hi) << 16);
}

// ---------------------------------------------------------------------------
// Pre-kernel 1: dense bf16 weights from circulant generators, in WMMA
// B-fragment order: blob[(f*128+q)*512 + L*16 + e], W[o,k]=c[o>>8,k>>8,(o-k)&255]
// ---------------------------------------------------------------------------
__global__ __launch_bounds__(256)
void build_w_kernel(const float* __restrict__ c, unsigned short* __restrict__ w) {
    unsigned int gid  = blockIdx.x * 256u + threadIdx.x;
    unsigned int base = gid * 4u;
    unsigned int e4 = base & 15u;
    unsigned int L  = (base >> 4) & 31u;
    unsigned int q  = (base >> 9) & 127u;
    unsigned int f  = base >> 16;
    unsigned int o  = f * 16u + (L & 15u);
    unsigned int kb = q * 32u + (L >> 4) * 16u + e4;
    unsigned int j  = o >> 8;
    unsigned int oc = o & 255u;
    unsigned int k0 = kb, k1 = kb + 1u, k2 = kb + 2u, k3 = kb + 3u;
    float v0 = c[(j * 16u + (k0 >> 8)) * 256u + ((oc - (k0 & 255u)) & 255u)];
    float v1 = c[(j * 16u + (k1 >> 8)) * 256u + ((oc - (k1 & 255u)) & 255u)];
    float v2 = c[(j * 16u + (k2 >> 8)) * 256u + ((oc - (k2 & 255u)) & 255u)];
    float v3 = c[(j * 16u + (k3 >> 8)) * 256u + ((oc - (k3 & 255u)) & 255u)];
    ((uint2*)w)[gid] = make_uint2(packbf(v0, v1), packbf(v2, v3));
}

// ---------------------------------------------------------------------------
// Pre-kernel 2: x fp32 -> bf16, row-major (one-time; keeps GEMM loop VALU-free)
// ---------------------------------------------------------------------------
__global__ __launch_bounds__(256)
void cvt_x_kernel(const float* __restrict__ x, unsigned short* __restrict__ xb) {
    unsigned int gid = blockIdx.x * 256u + threadIdx.x;
    const float4* src = (const float4*)x + (size_t)gid * 2u;
    float4 a = src[0], b = src[1];
    uint4 pk;
    pk.x = packbf(a.x, a.y);
    pk.y = packbf(a.z, a.w);
    pk.z = packbf(b.x, b.y);
    pk.w = packbf(b.z, b.w);
    ((uint4*)xb)[gid] = pk;
}

// ---------------------------------------------------------------------------
// Main GEMM: out = x_bf16 @ W^T + bias
// 256 thr (8 wave32), block tile 128(M) x 256(N), K-step 64; wave tile 64x64.
// A tile double-buffered in LDS, rows padded to 144 B by the TDM pad feature
// (conflict-free 16B fragment reads). Fallback = manual b128 copies.
// ---------------------------------------------------------------------------
#define MT   128
#define NT   256
#define KT   64
#define LDA  144
#define TILE_B (MT * LDA)   // 18432 bytes per LDS buffer

__device__ __forceinline__ unsigned lds_addr_of(const void* p) {
    return (unsigned)(uintptr_t)(const __attribute__((address_space(3))) void*)p;
}

#if TDM_PATH
__device__ __forceinline__ void tdm_load_tile(const unsigned short* gsrc,
                                              unsigned ldsByteAddr, int M, int K) {
    unsigned long long ga = (unsigned long long)(uintptr_t)gsrc;
    u32x4 g0 = { 1u,                                   // count=1 (valid D#)
                 ldsByteAddr,                          // lds_addr
                 (unsigned)ga,                         // global_addr[31:0]
                 (unsigned)((ga >> 32) & 0x01FFFFFFull) | (2u << 30) }; // addr[56:32] | type=2
    // data_size=2B(code1), pad_enable=1, pad_interval=32DW(code4), pad_amount=4DW(code3)
    i32x8 g1 = { (int)0x07110000u,
                 (int)(((unsigned)K & 0xFFFFu) << 16),            // tensor_dim0 lo16
                 (int)((((unsigned)K >> 16) & 0xFFFFu) |
                       (((unsigned)M & 0xFFFFu) << 16)),          // dim0 hi | dim1 lo
                 (int)((((unsigned)M >> 16) & 0xFFFFu) |
                       ((unsigned)KT << 16)),                     // dim1 hi | tile_dim0=64
                 (int)MT,                                         // tile_dim1=128, tile_dim2=0
                 (int)K,                                          // tensor_dim0_stride lo32
                 0, 0 };
    i32x4 z4 = { 0, 0, 0, 0 };
#if __clang_major__ >= 23
    i32x8 z8 = { 0, 0, 0, 0, 0, 0, 0, 0 };
    __builtin_amdgcn_tensor_load_to_lds(g0, g1, z4, z4, z8, 0);
#else
    __builtin_amdgcn_tensor_load_to_lds(g0, g1, z4, z4, 0);
#endif
}
#endif

__global__ __launch_bounds__(256)
void bcgemm_kernel(const unsigned short* __restrict__ xb,
                   const unsigned short* __restrict__ w,
                   const float* __restrict__ bias,
                   float* __restrict__ out,
                   int M, int K, int N) {
    __shared__ unsigned char lds[2 * TILE_B];

    const unsigned int tid  = threadIdx.x;
    const unsigned int lane = tid & 31u;
    const unsigned int wave = tid >> 5;
    const unsigned int wm   = wave >> 2;    // 0..1 -> M offset 64*wm
    const unsigned int wn   = wave & 3u;    // 0..3 -> N offset 64*wn
    const unsigned int l15  = lane & 15u;
    const unsigned int h    = lane >> 4;

    const unsigned int m0 = blockIdx.y * MT;
    const unsigned int n0 = blockIdx.x * NT;

    v8f acc[4][4];
    const v8f vzero = {0.f, 0.f, 0.f, 0.f, 0.f, 0.f, 0.f, 0.f};
#pragma unroll
    for (int mi = 0; mi < 4; ++mi)
#pragma unroll
        for (int nf = 0; nf < 4; ++nf) acc[mi][nf] = vzero;

    float bv[4];
#pragma unroll
    for (int nf = 0; nf < 4; ++nf)
        bv[nf] = bias[n0 + wn * 64u + (unsigned int)nf * 16u + l15];

    const unsigned int steps = (unsigned int)K / KT;
    unsigned int buf = 0;
    const unsigned ldsBase = lds_addr_of(&lds[0]);
    (void)ldsBase;

#if TDM_PATH
    if (wave == 0)
        tdm_load_tile(xb + (size_t)m0 * (size_t)K, ldsBase, M, K);
#else
    // manual staging: thread -> (row sr, 64B half sh) of the 128x64 bf16 tile
    const unsigned int sr = tid >> 1;
    const unsigned int sh = tid & 1u;
    const unsigned short* xrow = xb + (size_t)(m0 + sr) * (size_t)K + sh * 32u;
    uint4 rx[4];
#pragma unroll
    for (int i = 0; i < 4; ++i) rx[i] = ((const uint4*)xrow)[i];
#endif

    for (unsigned int s = 0; s < steps; ++s) {
#if TDM_PATH
        if (wave == 0) __builtin_amdgcn_s_wait_tensorcnt(0);  // tile s landed
        __syncthreads();
        if (wave == 0 && s + 1 < steps)                       // overlap tile s+1
            tdm_load_tile(xb + (size_t)m0 * (size_t)K + (size_t)(s + 1) * KT,
                          ldsBase + (buf ^ 1u) * TILE_B, M, K);
#else
        unsigned char* abase = lds + buf * TILE_B + sr * LDA + sh * 64u;
#pragma unroll
        for (int i = 0; i < 4; ++i) *((uint4*)(abase + i * 16)) = rx[i];
        if (s + 1 < steps) {
            const uint4* src = (const uint4*)(xrow + (size_t)(s + 1) * KT);
#pragma unroll
            for (int i = 0; i < 4; ++i) rx[i] = src[i];
        }
        __syncthreads();
#endif

#pragma unroll
        for (int kh = 0; kh < 2; ++kh) {
            const int kk = kh * 32;
            const unsigned int qk = s * 2u + (unsigned int)kh;

            // ---- 4 B fragments for this K sub-step (global, L2-hot blob) ----
            Frag16 bfr[4];
#pragma unroll
            for (int nf = 0; nf < 4; ++nf) {
                const unsigned int f = (n0 + wn * 64u + (unsigned int)nf * 16u) >> 4;
                const unsigned char* bb = (const unsigned char*)w +
                    ((size_t)f * 128u + qk) * 1024u + lane * 32u;
                bfr[nf].q[0] = *((const uint4*)bb);
                bfr[nf].q[1] = *((const uint4*)(bb + 16));
                if (s + 1 < steps)
                    __builtin_prefetch(bb + 2048, 0, 1);
            }
            // ---- 4 A fragments up front (distinct regs, overlapped DS) ----
            Frag16 afr[4];
#pragma unroll
            for (int mi = 0; mi < 4; ++mi) {
                const unsigned int mloc = wm * 64u + (unsigned int)mi * 16u + l15;
                const unsigned char* ab =
                    lds + buf * TILE_B + mloc * LDA + (unsigned int)kk * 2u + h * 32u;
                afr[mi].q[0] = *((const uint4*)ab);         // K = kk + 16h ..
                afr[mi].q[1] = *((const uint4*)(ab + 16));  // K = kk + 16h + 8 ..
            }
            // ---- 16 WMMAs: each A fragment feeds 4, each B fragment feeds 4 ----
#pragma unroll
            for (int mi = 0; mi < 4; ++mi)
#pragma unroll
                for (int nf = 0; nf < 4; ++nf)
                    acc[mi][nf] = __builtin_amdgcn_wmma_f32_16x16x32_bf16(
                        false, afr[mi].v, false, bfr[nf].v,
                        (short)0, acc[mi][nf], false, false);
        }
        buf ^= 1u;
    }

    // epilogue: C layout lane=col(l15), VGPR r = row h*8+r
#pragma unroll
    for (int mi = 0; mi < 4; ++mi) {
        const unsigned int m = m0 + wm * 64u + (unsigned int)mi * 16u + h * 8u;
#pragma unroll
        for (int nf = 0; nf < 4; ++nf) {
            const unsigned int n = n0 + wn * 64u + (unsigned int)nf * 16u + l15;
            float* op = out + (size_t)m * (size_t)N + n;
#pragma unroll
            for (int r = 0; r < 8; ++r)
                op[(size_t)r * (size_t)N] = acc[mi][nf][r] + bv[nf];
        }
    }
}

extern "C" void kernel_launch(void* const* d_in, const int* in_sizes, int n_in,
                              void* d_out, int out_size, void* d_ws, size_t ws_size,
                              hipStream_t stream) {
    const float* x    = (const float*)d_in[0];   // (4,2048,4096) fp32
    const float* c    = (const float*)d_in[1];   // (16,16,256)   fp32
    const float* bias = (const float*)d_in[2];   // (4096,)       fp32
    float* out = (float*)d_out;

    const int N = 16 * 256;
    const int K = 16 * 256;
    const int M = in_sizes[0] / K;               // 8192

    unsigned short* w  = (unsigned short*)d_ws;                       // 33.5 MB
    unsigned short* xb = (unsigned short*)((char*)d_ws + (size_t)N * K * 2); // 64 MB

    build_w_kernel<<<(N * K / 4) / 256, 256, 0, stream>>>(c, w);
    cvt_x_kernel<<<((size_t)M * K / 8) / 256, 256, 0, stream>>>(x, xb);

    dim3 grid(N / NT, M / MT);                   // (16, 64)
    bcgemm_kernel<<<grid, dim3(256), 0, stream>>>(xb, w, bias, out, M, K, N);
}
